// TBVFI_1529008357734
// MI455X (gfx1250) — matrix operand order repaired and checked by reading, not compiled, not run on hardware
//
#include <hip/hip_runtime.h>

typedef __attribute__((ext_vector_type(16))) _Float16 v16h;
typedef __attribute__((ext_vector_type(8)))  _Float16 v8h;
typedef __attribute__((ext_vector_type(8)))  float    v8f;

#define WGSIZE 256

static inline int cdiv_h(int a, int b) { return (a + b - 1) / b; }

// ---------------------------------------------------------------------------
// Conv2d as implicit GEMM on WMMA.
// GEMM: M = Ho*Wo pixels (tile 64/block), N = Cout (tile 128/block),
//       K = Cin*KH*KW stepped by 32 (fp16 fragments, f32 accumulate).
// 8 waves/block as 2 (pixel) x 4 (oc) subtiles; each wave computes
// 32px x 32oc = four 16x16 D tiles -> 4 WMMAs per K step from 8 ds_load_b128
// (2 LDS loads per WMMA). LDS double-buffered (stage s+1 while computing s).
// Row stride 40 halves = 80B: 16B-aligned for b128 fragment reads and
// bank-conflict-free (20-bank stride, 16 lanes -> 16 distinct banks).
// Zero division/branching in the K loop: rr->(kh,kw) comes from a tiny LDS
// table built once; B addressing is a pure arithmetic progression; (cin,rr)
// advance incrementally. Epilogue fuses bias and optional residual add.
// ---------------------------------------------------------------------------
__global__ __launch_bounds__(256)
void conv2d_wmma_kernel(const float* __restrict__ in, const float* __restrict__ w,
                        const float* __restrict__ bias, const float* __restrict__ resid,
                        float* __restrict__ out,
                        int Cin, int H, int W, int Cout, int Ho, int Wo,
                        int KH, int KW, int stride, int pad)
{
  __shared__ __align__(16) _Float16 lA[2][64][40];    // pixels x k
  __shared__ __align__(16) _Float16 lB[2][128][40];   // oc x k
  __shared__ int s_khw[16];                           // rr -> (kh<<16)|kw

  const int tid    = threadIdx.x;
  const int lane   = tid & 31;
  const int wid    = tid >> 5;
  const int wavep  = wid >> 2;       // 0..1 : 32-pixel subtile
  const int waveoc = wid & 3;        // 0..3 : 32-oc subtile
  const int P      = Ho * Wo;
  const int pix0   = blockIdx.x * 64;
  const int oc0    = blockIdx.y * 128;
  const int KWH    = KH * KW;
  const int Ktot   = Cin * KWH;
  const int HWin   = H * W;

  if (tid < KWH) {
    int kh = tid / KW;
    s_khw[tid] = (kh << 16) | (tid - kh * KW);
  }

  // ---- per-thread staging state (fixed across the K loop) ----
  const int kk      = tid & 31;      // k offset within chunk (fixed)
  const int rowbase = tid >> 5;      // stages A rows rowbase+8i, B rows rowbase+8j
  int  iy0a[8], ix0a[8];
  int  pvm = 0;                      // A-pixel validity bitmask
#pragma unroll
  for (int i = 0; i < 8; ++i) {
    int pixel = pix0 + rowbase + 8 * i;
    bool v = pixel < P;
    int pp = v ? pixel : 0;
    int oy = pp / Wo, ox = pp - oy * Wo;
    iy0a[i] = oy * stride - pad;
    ix0a[i] = ox * stride - pad;
    if (v) pvm |= 1 << i;
  }
  int bvm = 0;                       // B-oc validity bitmask (16 rows)
#pragma unroll
  for (int j = 0; j < 16; ++j)
    if (oc0 + rowbase + 8 * j < Cout) bvm |= 1 << j;
  const long dK   = 8L * Ktot;       // B row-to-row weight stride
  long woff = (long)(oc0 + rowbase) * Ktot + kk;   // running weight offset

  // incremental k -> (cin, rr) decomposition
  const int d32 = 32 / KWH;
  const int m32 = 32 - d32 * KWH;
  int cin = kk / KWH;                // one division, prologue only
  int rr  = kk - cin * KWH;
  int kg  = kk;

  __syncthreads();   // s_khw visible

  auto STAGE = [&](int buf, int scin, int srr, int skg, long swoff) {
    const bool kvalid = skg < Ktot;
    const int  pk = s_khw[srr];
    const int  kh = pk >> 16, kw = pk & 0xffff;
#pragma unroll
    for (int i = 0; i < 8; ++i) {
      float va = 0.0f;
      int yy = iy0a[i] + kh, xx = ix0a[i] + kw;
      if (kvalid && ((pvm >> i) & 1) && yy >= 0 && yy < H && xx >= 0 && xx < W)
        va = in[(long)scin * HWin + (long)yy * W + xx];
      lA[buf][rowbase + 8 * i][kk] = (_Float16)va;
    }
    const float* wp = w + swoff;
#pragma unroll
    for (int j = 0; j < 16; ++j) {
      float vb = 0.0f;
      if (kvalid && ((bvm >> j) & 1)) vb = wp[0];
      lB[buf][rowbase + 8 * j][kk] = (_Float16)vb;
      wp += dK;
    }
    // speculative prefetch of next weight tile (global_prefetch_b8)
    if ((bvm & 1) && (skg + 32) < Ktot)
      __builtin_prefetch((const void*)(w + swoff + 32), 0, 0);
  };

  const int nsteps = (Ktot + 31) >> 5;

  STAGE(0, cin, rr, kg, woff);
  cin += d32; rr += m32; if (rr >= KWH) { rr -= KWH; ++cin; } kg += 32; woff += 32;
  __syncthreads();

  v8f acc00 = {}, acc01 = {}, acc10 = {}, acc11 = {};
  const int m  = lane & 15;
  const int hi = lane >> 4;

  for (int s = 0; s < nsteps; ++s) {
    int buf = s & 1;
    if (s + 1 < nsteps) {
      STAGE(buf ^ 1, cin, rr, kg, woff);
      cin += d32; rr += m32; if (rr >= KWH) { rr -= KWH; ++cin; } kg += 32; woff += 32;
    }
    // ---- fragments via contiguous 16B LDS reads (documented wave32 layouts) ----
    union U { v16h v; v8h h[2]; };
    U A0, A1, B0, B1;
    const _Float16* ar0 = &lA[buf][wavep * 32 + m][0];
    const _Float16* ar1 = &lA[buf][wavep * 32 + 16 + m][0];
    A0.h[0] = *(const v8h*)(ar0 + hi * 8);
    A0.h[1] = *(const v8h*)(ar0 + 16 + hi * 8);
    A1.h[0] = *(const v8h*)(ar1 + hi * 8);
    A1.h[1] = *(const v8h*)(ar1 + 16 + hi * 8);
    const _Float16* br0 = &lB[buf][waveoc * 32 + m][0];
    const _Float16* br1 = &lB[buf][waveoc * 32 + 16 + m][0];
    B0.h[0] = *(const v8h*)(br0 + hi * 8);
    B0.h[1] = *(const v8h*)(br0 + 16 + hi * 8);
    B1.h[0] = *(const v8h*)(br1 + hi * 8);
    B1.h[1] = *(const v8h*)(br1 + 16 + hi * 8);
    acc00 = __builtin_amdgcn_wmma_f32_16x16x32_f16(false, A0.v, false, B0.v,
                                                   (short)0, acc00, false, false);
    acc01 = __builtin_amdgcn_wmma_f32_16x16x32_f16(false, A0.v, false, B1.v,
                                                   (short)0, acc01, false, false);
    acc10 = __builtin_amdgcn_wmma_f32_16x16x32_f16(false, A1.v, false, B0.v,
                                                   (short)0, acc10, false, false);
    acc11 = __builtin_amdgcn_wmma_f32_16x16x32_f16(false, A1.v, false, B1.v,
                                                   (short)0, acc11, false, false);
    __syncthreads();
  }

  // ---- epilogue: D layout lane n = L&15, row m = r + (L>=16)*8 ----
  const int nn = lane & 15;
#pragma unroll
  for (int pa = 0; pa < 2; ++pa) {
    int base = pix0 + wavep * 32 + pa * 16 + hi * 8;
    int oy = base / Wo, ox = base - oy * Wo;
#pragma unroll
    for (int r = 0; r < 8; ++r) {
      int pixel = base + r;
      if (pixel < P) {
#pragma unroll
        for (int tb = 0; tb < 2; ++tb) {
          int oc = oc0 + waveoc * 32 + tb * 16 + nn;
          if (oc < Cout) {
            float av = pa == 0 ? (tb == 0 ? acc00[r] : acc01[r])
                               : (tb == 0 ? acc10[r] : acc11[r]);
            long idx = ((long)oc * Ho + oy) * Wo + ox;
            float v = av + bias[oc];
            if (resid) v += resid[idx];
            out[idx] = v;
          }
        }
      }
      ++ox; if (ox == Wo) { ox = 0; ++oy; }
    }
  }
}

// ---------------------------------------------------------------------------
// Fused GroupNorm(32 groups) + swish. One block per group (B=1).
// ---------------------------------------------------------------------------
__global__ __launch_bounds__(256)
void gn_swish_kernel(const float* __restrict__ x, const float* __restrict__ g,
                     const float* __restrict__ b, float* __restrict__ y,
                     int C, int HW, int cpg)
{
  const int grp = blockIdx.x;
  const int c0  = grp * cpg;
  const long n  = (long)cpg * HW;
  const float* xs = x + (long)c0 * HW;
  float s1 = 0.0f, s2 = 0.0f;
  for (long i = threadIdx.x; i < n; i += blockDim.x) {
    float v = xs[i];
    s1 += v; s2 += v * v;
  }
  __shared__ float sh1[256], sh2[256];
  sh1[threadIdx.x] = s1; sh2[threadIdx.x] = s2;
  __syncthreads();
  for (int off = 128; off > 0; off >>= 1) {
    if (threadIdx.x < off) {
      sh1[threadIdx.x] += sh1[threadIdx.x + off];
      sh2[threadIdx.x] += sh2[threadIdx.x + off];
    }
    __syncthreads();
  }
  float mu  = sh1[0] / (float)n;
  float var = sh2[0] / (float)n - mu * mu;
  float inv = rsqrtf(var + 1e-6f);
  for (long i = threadIdx.x; i < n; i += blockDim.x) {
    int c = c0 + (int)(i / HW);
    float v = (xs[i] - mu) * inv * g[c] + b[c];
    y[(long)c0 * HW + i] = v / (1.0f + __expf(-v));
  }
}

__global__ void fill_kernel(float* p, long n, float v)
{
  long i = (long)blockIdx.x * blockDim.x + threadIdx.x;
  if (i < n) p[i] = v;
}

// ---------------------------------------------------------------------------
// softsplat_avg: scatter (bilinear forward splat with atomics) + normalize.
// acc layout: (B, C+1, HW); channel C is the weight accumulator.
// ---------------------------------------------------------------------------
__global__ void softsplat_scatter_kernel(const float* __restrict__ inp,
                                         const float* __restrict__ flow,
                                         float* __restrict__ acc,
                                         int B, int C, int H, int W)
{
  long idx = (long)blockIdx.x * blockDim.x + threadIdx.x;
  const int HW = H * W;
  if (idx >= (long)B * HW) return;
  int bb = (int)(idx / HW);
  int p  = (int)(idx - (long)bb * HW);
  int yy = p / W, xx = p - yy * W;
  const float* fl = flow + (long)bb * 2 * HW;
  float fx = xx + fl[p];
  float fy = yy + fl[HW + p];
  float x0 = floorf(fx), y0 = floorf(fy);
  float* ab = acc + (long)bb * (C + 1) * HW;
  const float* ib = inp + (long)bb * C * HW;
  for (int d = 0; d < 4; ++d) {
    int dx = d & 1, dy = d >> 1;
    float xi = x0 + dx, yi = y0 + dy;
    float wgt = (1.0f - fabsf(fx - xi)) * (1.0f - fabsf(fy - yi));
    if (xi >= 0.0f && xi < (float)W && yi >= 0.0f && yi < (float)H) {
      int ti = (int)yi * W + (int)xi;
      for (int c = 0; c < C; ++c)
        atomicAdd(&ab[(long)c * HW + ti], ib[(long)c * HW + p] * wgt);
      atomicAdd(&ab[(long)C * HW + ti], wgt);
    }
  }
}

__global__ void softsplat_norm_kernel(const float* __restrict__ acc,
                                      float* __restrict__ out,
                                      int B, int C, int HW, float sign)
{
  long idx = (long)blockIdx.x * blockDim.x + threadIdx.x;
  if (idx >= (long)B * C * HW) return;
  int bb = (int)(idx / ((long)C * HW));
  long r = idx - (long)bb * C * HW;
  int p  = (int)(r % HW);
  float nm = acc[((long)bb * (C + 1) + C) * HW + p];
  if (nm == 0.0f) nm = 1.0f;
  out[idx] = sign * acc[(long)bb * (C + 1) * HW + r] / nm;
}

// 2x2 mean downsample with scale (resize_half_bilinear * scale)
__global__ void avgpool2_scale_kernel(const float* __restrict__ x, float* __restrict__ y,
                                      int C, int H, int W, float s)
{
  int Ho = H / 2, Wo = W / 2;
  long n = (long)C * Ho * Wo;
  long i = (long)blockIdx.x * blockDim.x + threadIdx.x;
  if (i >= n) return;
  int c  = (int)(i / ((long)Ho * Wo));
  long r = i - (long)c * Ho * Wo;
  int yy = (int)(r / Wo), xx = (int)(r - (long)yy * Wo);
  const float* xp = x + ((long)c * H + 2 * yy) * W + 2 * xx;
  y[i] = 0.25f * (xp[0] + xp[1] + xp[W] + xp[W + 1]) * s;
}

// strided ::2 subsample with scale
__global__ void subsample2_scale_kernel(const float* __restrict__ x, float* __restrict__ y,
                                        int C, int H, int W, float s)
{
  int Ho = H / 2, Wo = W / 2;
  long n = (long)C * Ho * Wo;
  long i = (long)blockIdx.x * blockDim.x + threadIdx.x;
  if (i >= n) return;
  int c  = (int)(i / ((long)Ho * Wo));
  long r = i - (long)c * Ho * Wo;
  int yy = (int)(r / Wo), xx = (int)(r - (long)yy * Wo);
  y[i] = x[((long)c * H + 2 * yy) * W + 2 * xx] * s;
}

// 2x nearest upsample
__global__ void upsample2x_kernel(const float* __restrict__ x, float* __restrict__ y,
                                  int C, int H, int W)
{
  int H2 = 2 * H, W2 = 2 * W;
  long n = (long)C * H2 * W2;
  long i = (long)blockIdx.x * blockDim.x + threadIdx.x;
  if (i >= n) return;
  int c  = (int)(i / ((long)H2 * W2));
  long r = i - (long)c * H2 * W2;
  int yy = (int)(r / W2), xx = (int)(r - (long)yy * W2);
  y[i] = x[((long)c * H + (yy >> 1)) * W + (xx >> 1)];
}

// warp_nearest: gather img (C,Hi,Wi) at flow grid (2,H,W) -> (C,H,W)
__global__ void warp_nearest_kernel(const float* __restrict__ img,
                                    const float* __restrict__ flow,
                                    float* __restrict__ out,
                                    int C, int Hi, int Wi, int H, int W)
{
  int p = blockIdx.x * blockDim.x + threadIdx.x;
  if (p >= H * W) return;
  int yy = p / W, xx = p - yy * W;
  float gx = (-1.0f + 2.0f * xx / (float)(W - 1)) + flow[p] * (2.0f / (float)(W - 1));
  float gy = (-1.0f + 2.0f * yy / (float)(H - 1)) + flow[H * W + p] * (2.0f / (float)(H - 1));
  int ix = (int)floorf((gx + 1.0f) * 0.5f * (float)(Wi - 1) + 0.5f);
  int iy = (int)floorf((gy + 1.0f) * 0.5f * (float)(Hi - 1) + 0.5f);
  ix = min(max(ix, 0), Wi - 1);
  iy = min(max(iy, 0), Hi - 1);
  for (int c = 0; c < C; ++c)
    out[(long)c * H * W + p] = img[((long)c * Hi + iy) * Wi + ix];
}

// fused unfold(feat,2,2,1,1) + warp_nearest: out (C*4,H,W), virtual img (Hf-1,Wf-1)
__global__ void warp_unfold_kernel(const float* __restrict__ feat,
                                   const float* __restrict__ flow,
                                   float* __restrict__ out,
                                   int C, int Hf, int Wf, int H, int W)
{
  int p = blockIdx.x * blockDim.x + threadIdx.x;
  if (p >= H * W) return;
  int Hi = Hf - 1, Wi = Wf - 1;
  int yy = p / W, xx = p - yy * W;
  float gx = (-1.0f + 2.0f * xx / (float)(W - 1)) + flow[p] * (2.0f / (float)(W - 1));
  float gy = (-1.0f + 2.0f * yy / (float)(H - 1)) + flow[H * W + p] * (2.0f / (float)(H - 1));
  int ix = (int)floorf((gx + 1.0f) * 0.5f * (float)(Wi - 1) + 0.5f);
  int iy = (int)floorf((gy + 1.0f) * 0.5f * (float)(Hi - 1) + 0.5f);
  ix = min(max(ix, 0), Wi - 1);
  iy = min(max(iy, 0), Hi - 1);
  for (int c = 0; c < C; ++c)
    for (int dy = 0; dy < 2; ++dy)
      for (int dx = 0; dx < 2; ++dx)
        out[((long)(c * 2 + dy) * 2 + dx) * H * W + p] =
            feat[((long)c * Hf + iy + dy) * Wf + ix + dx];
}

// fold_nonoverlap with k=2: in (C*4, nh, nw) -> out (C, 2nh, 2nw)
__global__ void fold2_kernel(const float* __restrict__ in, float* __restrict__ out,
                             int C, int nh, int nw)
{
  int H = 2 * nh, W = 2 * nw;
  long n = (long)C * H * W;
  long i = (long)blockIdx.x * blockDim.x + threadIdx.x;
  if (i >= n) return;
  int c  = (int)(i / ((long)H * W));
  long r = i - (long)c * H * W;
  int yy = (int)(r / W), xx = (int)(r - (long)yy * W);
  int ki = yy & 1, kj = xx & 1;
  out[i] = in[((long)(c * 4 + ki * 2 + kj) * nh + (yy >> 1)) * nw + (xx >> 1)];
}

// ---------------------------------------------------------------------------
// Texture search kernels. Q/K at 56x56, SS=2 -> nh=nw=28, L=784.
// ---------------------------------------------------------------------------
__global__ __launch_bounds__(128)
void tex_sq_kernel(const float* __restrict__ Q, float* __restrict__ sQ,
                   int C, int Hq, int Wq)
{
  int nw = Wq / 2;
  int l  = blockIdx.x;
  int ly = l / nw, lx = l - ly * nw;
  const int HW = Hq * Wq;
  float ss = 0.0f;
  for (int i = threadIdx.x; i < C * 4; i += blockDim.x) {
    int c = i >> 2, pos = i & 3;
    int yy = 2 * ly + (pos >> 1), xx = 2 * lx + (pos & 1);
    float v = Q[(long)c * HW + yy * Wq + xx];
    ss += v * v;
  }
  __shared__ float sh[128];
  sh[threadIdx.x] = ss;
  __syncthreads();
  for (int off = 64; off > 0; off >>= 1) {
    if (threadIdx.x < off) sh[threadIdx.x] += sh[threadIdx.x + off];
    __syncthreads();
  }
  float inv = 1.0f / fmaxf(sqrtf(sh[0]), 1e-12f);
  for (int c = threadIdx.x; c < C; c += blockDim.x) {
    const float* qp = Q + (long)c * HW + (2 * ly) * Wq + 2 * lx;
    float m = qp[0] + qp[1] + qp[Wq] + qp[Wq + 1];
    sQ[(long)l * C + c] = 0.25f * m * inv;
  }
}

// sK for frame f in {0,1}; output sK[(f*C + c)*L + l]
__global__ __launch_bounds__(128)
void tex_sk_kernel(const float* __restrict__ K0, const float* __restrict__ K1,
                   float* __restrict__ sK, int C, int Hk, int Wk)
{
  int nw = Wk / 2, nh = Hk / 2;
  int L  = nh * nw;
  int l  = blockIdx.x;
  int f  = blockIdx.y;
  const float* K = f ? K1 : K0;
  int ly = l / nw, lx = l - ly * nw;
  const int HW = Hk * Wk;
  float ss = 0.0f;
  for (int i = threadIdx.x; i < C * 4; i += blockDim.x) {
    int c = i >> 2, pos = i & 3;
    int yy = 2 * ly + (pos >> 1), xx = 2 * lx + (pos & 1);
    float v = K[(long)c * HW + yy * Wk + xx];
    ss += v * v;
  }
  __shared__ float sh[128];
  sh[threadIdx.x] = ss;
  __syncthreads();
  for (int off = 64; off > 0; off >>= 1) {
    if (threadIdx.x < off) sh[threadIdx.x] += sh[threadIdx.x + off];
    __syncthreads();
  }
  float inv = 1.0f / fmaxf(sqrtf(sh[0]), 1e-12f);
  for (int c = threadIdx.x; c < C; c += blockDim.x) {
    const float* kp = K + (long)c * HW + (2 * ly) * Wk + 2 * lx;
    float m = kp[0] + kp[1] + kp[Wk] + kp[Wk + 1];
    sK[((long)f * C + c) * L + l] = 0.25f * m * inv;
  }
}

// per-l argmax over 18 candidates (2 frames x 3x3 neighborhood, zero-padded)
__global__ __launch_bounds__(128)
void tex_argmax_kernel(const float* __restrict__ sK, const float* __restrict__ sQ,
                       int* __restrict__ best, int C, int nh, int nw)
{
  int L  = nh * nw;
  int l  = blockIdx.x;
  int ly = l / nw, lx = l - ly * nw;
  __shared__ float sd[18 * 128];
  int t = threadIdx.x;
  float q = 0.0f;
  int c = t;
  bool has = (c < C);
  if (has) q = sQ[(long)l * C + c];
#pragma unroll
  for (int k = 0; k < 18; ++k) {
    float v = 0.0f;
    if (has) {
      int f = k / 9, o = k - f * 9;
      int dy = o / 3, dx = o - dy * 3;
      int py = ly - 1 + dy, px = lx - 1 + dx;
      if (py >= 0 && py < nh && px >= 0 && px < nw)
        v = sK[((long)f * C + c) * L + py * nw + px] * q;
    }
    sd[k * 128 + t] = v;
  }
  __syncthreads();
  for (int off = 64; off > 0; off >>= 1) {
    if (t < off)
      for (int k = 0; k < 18; ++k) sd[k * 128 + t] += sd[k * 128 + t + off];
    __syncthreads();
  }
  if (t == 0) {
    int bk = 0;
    float bs = sd[0];
    for (int k = 1; k < 18; ++k)
      if (sd[k * 128] > bs) { bs = sd[k * 128]; bk = k; }
    int f = bk / 9, o = bk - f * 9;
    int dy = o / 3, dx = o - dy * 3;
    int py = ly + dy, px = lx + dx;      // padded coords 0..29
    best[l] = (f << 10) | (py << 5) | px;
  }
}

// gather 4x4 V patches at argmax winners + fold to (C, 4nh, 4nw)
__global__ void tex_gather_kernel(const int* __restrict__ best,
                                  const float* __restrict__ V0,
                                  const float* __restrict__ V1,
                                  float* __restrict__ out,
                                  int C, int nh, int Hv)
{
  int Wv = Hv;
  long n = (long)C * Hv * Wv;
  long i = (long)blockIdx.x * blockDim.x + threadIdx.x;
  if (i >= n) return;
  int c  = (int)(i / ((long)Hv * Wv));
  long r = i - (long)c * Hv * Wv;
  int yy = (int)(r / Wv), xx = (int)(r - (long)yy * Wv);
  int oh = yy >> 2, ki = yy & 3;
  int ow = xx >> 2, kj = xx & 3;
  int l  = oh * nh + ow;
  int e  = best[l];
  int f  = e >> 10, py = (e >> 5) & 31, px = e & 31;
  float v = 0.0f;
  if (py >= 1 && py <= nh && px >= 1 && px <= nh) {
    const float* V = f ? V1 : V0;
    v = V[((long)c * Hv + ((py - 1) * 4 + ki)) * Wv + ((px - 1) * 4 + kj)];
  }
  out[i] = v;
}

// ===========================================================================
// Host orchestration
// ===========================================================================
struct Cur { void* const* arr; int i; };
static const float* nxt(Cur& c) { return (const float*)c.arr[c.i++]; }

struct ConvP { const float* b; const float* w; };
static ConvP take_conv(Cur& c) { ConvP p; p.b = nxt(c); p.w = nxt(c); return p; }
struct GNP { const float* b; const float* g; };
static GNP take_gn(Cur& c) { GNP p; p.b = nxt(c); p.g = nxt(c); return p; }
struct ResP { ConvP c1, c2; GNP g1, g2; ConvP sk; int has_sk; };
static ResP take_res(Cur& c, int hs) {
  ResP r; r.c1 = take_conv(c); r.c2 = take_conv(c);
  r.g1 = take_gn(c); r.g2 = take_gn(c);
  r.has_sk = hs;
  if (hs) r.sk = take_conv(c); else { r.sk.b = nullptr; r.sk.w = nullptr; }
  return r;
}
struct StemP { ConvP c; ConvP d; ResP r; };
static StemP take_stem(Cur& c) {  // sorted keys: c, d, r
  StemP s; s.c = take_conv(c); s.d = take_conv(c); s.r = take_res(c, 0); return s;
}
struct LiteP { ResP d0r, d1r; ConvP d1u; ResP d2r; ConvP d2u;
               ConvP e1d; ResP e1r; ConvP e2d; ResP e2r; ConvP e3d; ResP e3r; ResP e4r; };
static LiteP take_lite(Cur& c) {  // sorted keys
  LiteP L;
  L.d0r = take_res(c, 1); L.d1r = take_res(c, 1); L.d1u = take_conv(c);
  L.d2r = take_res(c, 0); L.d2u = take_conv(c);
  L.e1d = take_conv(c); L.e1r = take_res(c, 1);
  L.e2d = take_conv(c); L.e2r = take_res(c, 1);
  L.e3d = take_conv(c); L.e3r = take_res(c, 1);
  L.e4r = take_res(c, 0);
  return L;
}
struct UnetP {
  ResP d0a, d0b, d1a, d1b; ConvP d1u; ResP d2a, d2b; ConvP d2u;
  ResP d3a, d3b; ConvP d3u; ResP d4a, d4b; ConvP d4u;
  ConvP dfc; GNP dfgn; ResP dha, dhb; ConvP dhc;
  ResP e1a, e1b; ConvP e1d; ResP e2a, e2b; ConvP e2d;
  ResP e3a, e3b; ConvP e3d; ResP e4a, e4b; ResP efa, efb; ConvP efc; GNP efgn;
};
static UnetP take_unet(Cur& c) {  // sorted keys
  UnetP U;
  U.d0a = take_res(c, 0); U.d0b = take_res(c, 0);
  U.d1a = take_res(c, 1); U.d1b = take_res(c, 0); U.d1u = take_conv(c);
  U.d2a = take_res(c, 1); U.d2b = take_res(c, 0); U.d2u = take_conv(c);
  U.d3a = take_res(c, 1); U.d3b = take_res(c, 0); U.d3u = take_conv(c);
  U.d4a = take_res(c, 1); U.d4b = take_res(c, 0); U.d4u = take_conv(c);
  U.dfc = take_conv(c); U.dfgn = take_gn(c);
  U.dha = take_res(c, 0); U.dhb = take_res(c, 0); U.dhc = take_conv(c);
  U.e1a = take_res(c, 0); U.e1b = take_res(c, 0); U.e1d = take_conv(c);
  U.e2a = take_res(c, 1); U.e2b = take_res(c, 0); U.e2d = take_conv(c);
  U.e3a = take_res(c, 0); U.e3b = take_res(c, 0); U.e3d = take_conv(c);
  U.e4a = take_res(c, 1); U.e4b = take_res(c, 0);
  U.efa = take_res(c, 0); U.efb = take_res(c, 0);
  U.efc = take_conv(c); U.efgn = take_gn(c);
  return U;
}

struct Ctx {
  hipStream_t st;
  char* base; size_t off; size_t cap;
  float *sA, *sB, *sC;   // reusable scratch slabs (128*224*224 floats each)
};
static float* wsalloc(Ctx& c, size_t elems) {
  size_t b = (elems * sizeof(float) + 255) & ~(size_t)255;
  float* p = (float*)(c.base + c.off);
  c.off += b;
  return p;
}
static void run_fill(Ctx& c, float* p, long n, float v) {
  fill_kernel<<<(int)((n + 255) / 256), 256, 0, c.st>>>(p, n, v);
}
static void run_gn(Ctx& c, const float* x, const GNP& g, float* y, int C, int HW) {
  gn_swish_kernel<<<32, 256, 0, c.st>>>(x, g.g, g.b, y, C, HW, C / 32);
}
static void run_conv(Ctx& c, const float* in, const ConvP& cw, const float* resid,
                     float* out, int Cin, int H, int W, int Cout,
                     int KH, int KW, int stride, int pad) {
  int Ho = (H + 2 * pad - KH) / stride + 1;
  int Wo = (W + 2 * pad - KW) / stride + 1;
  dim3 grid(cdiv_h(Ho * Wo, 64), cdiv_h(Cout, 128));
  conv2d_wmma_kernel<<<grid, 256, 0, c.st>>>(in, cw.w, cw.b, resid, out,
                                             Cin, H, W, Cout, Ho, Wo, KH, KW, stride, pad);
}
static float* res_block(Ctx& c, const float* x, int Cin, int Cout, int H, int W,
                        const ResP& r) {
  int HW = H * W;
  run_gn(c, x, r.g1, c.sA, Cin, HW);
  run_conv(c, c.sA, r.c1, nullptr, c.sB, Cin, H, W, Cout, 3, 3, 1, 1);
  run_gn(c, c.sB, r.g2, c.sA, Cout, HW);
  const float* resid = x;
  if (r.has_sk) {
    run_conv(c, x, r.sk, nullptr, c.sC, Cin, H, W, Cout, 1, 1, 1, 0);
    resid = c.sC;
  }
  float* out = wsalloc(c, (size_t)Cout * HW);
  run_conv(c, c.sA, r.c2, resid, out, Cout, H, W, Cout, 3, 3, 1, 1);
  return out;
}
static float* down_block(Ctx& c, const float* x, int C, int H, int W, const ConvP& d) {
  float* out = wsalloc(c, (size_t)C * (H / 2) * (W / 2));
  run_conv(c, x, d, nullptr, out, C, H, W, C, 3, 3, 2, 1);
  return out;
}
static float* up_block(Ctx& c, const float* x, int C, int H, int W, const ConvP& u) {
  long n = (long)C * 4 * H * W;
  upsample2x_kernel<<<(int)((n + 255) / 256), 256, 0, c.st>>>(x, c.sA, C, H, W);
  float* out = wsalloc(c, (size_t)n);
  run_conv(c, c.sA, u, nullptr, out, C, 2 * H, 2 * W, C, 3, 3, 1, 1);
  return out;
}
static float* cat2(Ctx& c, const float* a, int Ca, const float* b, int Cb, int HW) {
  float* o = wsalloc(c, (size_t)(Ca + Cb) * HW);
  hipMemcpyAsync(o, a, (size_t)Ca * HW * 4, hipMemcpyDeviceToDevice, c.st);
  hipMemcpyAsync(o + (long)Ca * HW, b, (size_t)Cb * HW * 4, hipMemcpyDeviceToDevice, c.st);
  return o;
}

static float* lite_unet(Ctx& c, const float* x, const LiteP& L) {
  float* r1 = res_block(c, x, 128, 32, 112, 112, L.e1r);
  float* o0 = down_block(c, r1, 32, 112, 112, L.e1d);      // 32@56
  float* r2 = res_block(c, o0, 32, 64, 56, 56, L.e2r);
  float* o1 = down_block(c, r2, 64, 56, 56, L.e2d);        // 64@28
  float* r3 = res_block(c, o1, 64, 128, 28, 28, L.e3r);
  float* o2 = down_block(c, r3, 128, 28, 28, L.e3d);       // 128@14
  float* t = res_block(c, o2, 128, 128, 14, 14, L.e4r);
  t = res_block(c, t, 128, 64, 14, 14, L.d0r);             // 64@14
  float* cc = cat2(c, t, 64, o2, 128, 14 * 14);            // 192@14
  t = res_block(c, cc, 192, 64, 14, 14, L.d1r);
  t = up_block(c, t, 64, 14, 14, L.d1u);                   // 64@28
  cc = cat2(c, t, 64, o1, 64, 28 * 28);                    // 128@28
  t = res_block(c, cc, 128, 128, 28, 28, L.d2r);
  t = up_block(c, t, 128, 28, 28, L.d2u);                  // 128@56
  return t;
}

static void unet_run(Ctx& c, const float* x, const UnetP& U, float* dout) {
  float* t = res_block(c, x, 128, 128, 112, 112, U.e1a);
  t = res_block(c, t, 128, 128, 112, 112, U.e1b);
  float* out0 = down_block(c, t, 128, 112, 112, U.e1d);    // 128@56
  t = res_block(c, out0, 128, 256, 56, 56, U.e2a);
  t = res_block(c, t, 256, 256, 56, 56, U.e2b);
  float* out1 = down_block(c, t, 256, 56, 56, U.e2d);      // 256@28
  t = res_block(c, out1, 256, 256, 28, 28, U.e3a);
  t = res_block(c, t, 256, 256, 28, 28, U.e3b);
  float* out2 = down_block(c, t, 256, 28, 28, U.e3d);      // 256@14
  t = res_block(c, out2, 256, 512, 14, 14, U.e4a);
  t = res_block(c, t, 512, 512, 14, 14, U.e4b);
  t = res_block(c, t, 512, 512, 14, 14, U.efa);
  t = res_block(c, t, 512, 512, 14, 14, U.efb);
  run_gn(c, t, U.efgn, c.sA, 512, 196);
  float* ef = wsalloc(c, (size_t)256 * 196);
  run_conv(c, c.sA, U.efc, nullptr, ef, 512, 14, 14, 256, 3, 3, 1, 1);
  float* dh = wsalloc(c, (size_t)512 * 196);
  run_conv(c, ef, U.dhc, nullptr, dh, 256, 14, 14, 512, 3, 3, 1, 1);
  t = res_block(c, dh, 512, 512, 14, 14, U.dha);
  t = res_block(c, t, 512, 512, 14, 14, U.dhb);
  t = res_block(c, t, 512, 512, 14, 14, U.d0a);
  t = res_block(c, t, 512, 512, 14, 14, U.d0b);
  float* cc = cat2(c, t, 512, out2, 256, 196);             // 768@14
  t = res_block(c, cc, 768, 256, 14, 14, U.d1a);
  t = res_block(c, t, 256, 256, 14, 14, U.d1b);
  t = up_block(c, t, 256, 14, 14, U.d1u);                  // 256@28
  cc = cat2(c, t, 256, out1, 256, 784);                    // 512@28
  t = res_block(c, cc, 512, 256, 28, 28, U.d2a);
  t = res_block(c, t, 256, 256, 28, 28, U.d2b);
  t = up_block(c, t, 256, 28, 28, U.d2u);                  // 256@56
  cc = cat2(c, t, 256, out0, 128, 3136);                   // 384@56
  t = res_block(c, cc, 384, 128, 56, 56, U.d3a);
  t = res_block(c, t, 128, 128, 56, 56, U.d3b);
  t = up_block(c, t, 128, 56, 56, U.d3u);                  // 128@112
  cc = cat2(c, t, 128, x, 128, 112 * 112);                 // 256@112
  t = res_block(c, cc, 256, 128, 112, 112, U.d4a);
  t = res_block(c, t, 128, 128, 112, 112, U.d4b);
  t = up_block(c, t, 128, 112, 112, U.d4u);                // 128@224
  run_gn(c, t, U.dfgn, c.sA, 128, 224 * 224);
  run_conv(c, c.sA, U.dfc, nullptr, dout, 128, 224, 224, 3, 3, 3, 1, 1);
}

extern "C" void kernel_launch(void* const* d_in, const int* in_sizes, int n_in,
                              void* d_out, int out_size, void* d_ws, size_t ws_size,
                              hipStream_t stream) {
  (void)in_sizes; (void)n_in; (void)out_size;
  const float* img0 = (const float*)d_in[0];   // (3,224,224)
  const float* img1 = (const float*)d_in[1];   // (3,224,224)
  const float* flow = (const float*)d_in[2];   // (4,224,224)

  Cur cur{d_in + 3, 0};
  StemP ehi = take_stem(cur);
  StemP ehw = take_stem(cur);
  LiteP lite = take_lite(cur);
  ResP mf1 = take_res(cur, 0);
  ResP mf2 = take_res(cur, 1);
  ResP mf3 = take_res(cur, 0);
  UnetP U = take_unet(cur);

  Ctx c;
  c.st = stream; c.base = (char*)d_ws; c.off = 0; c.cap = ws_size;
  const size_t SLAB = (size_t)128 * 224 * 224;     // largest transient tensor
  c.sA = wsalloc(c, SLAB);
  c.sB = wsalloc(c, SLAB);
  c.sC = wsalloc(c, SLAB);

  const int HW224 = 224 * 224, HW112 = 112 * 112;

  // ---- softmax splat of the two frames ----
  float* wf = wsalloc(c, (size_t)6 * HW224);
  float* accA = wsalloc(c, (size_t)4 * HW224);
  run_fill(c, accA, (long)4 * HW224, 0.0f);
  softsplat_scatter_kernel<<<cdiv_h(HW224, 256), 256, 0, stream>>>(img0, flow, accA, 1, 3, 224, 224);
  softsplat_norm_kernel<<<cdiv_h(3 * HW224, 256), 256, 0, stream>>>(accA, wf, 1, 3, HW224, 1.0f);
  run_fill(c, accA, (long)4 * HW224, 0.0f);
  softsplat_scatter_kernel<<<cdiv_h(HW224, 256), 256, 0, stream>>>(img1, flow + 2 * HW224, accA, 1, 3, 224, 224);
  softsplat_norm_kernel<<<cdiv_h(3 * HW224, 256), 256, 0, stream>>>(accA, wf + (long)3 * HW224, 1, 3, HW224, 1.0f);

  // ---- warped-feature stem (ehw) ----
  float* t0 = wsalloc(c, (size_t)128 * HW224);
  run_conv(c, wf, ehw.c, nullptr, t0, 6, 224, 224, 128, 3, 3, 1, 1);
  float* t1 = res_block(c, t0, 128, 128, 224, 224, ehw.r);
  float* wfeat = down_block(c, t1, 128, 224, 224, ehw.d);        // 128@112

  // ---- Q via lite U-Net ----
  float* Q = lite_unet(c, wfeat, lite);                          // 128@56

  // ---- image feature stems (shared ehi weights) ----
  float* f0c = wsalloc(c, (size_t)128 * HW224);
  run_conv(c, img0, ehi.c, nullptr, f0c, 3, 224, 224, 128, 3, 3, 1, 1);
  float* f0r = res_block(c, f0c, 128, 128, 224, 224, ehi.r);
  float* feat0 = down_block(c, f0r, 128, 224, 224, ehi.d);       // 128@112
  float* f1c = wsalloc(c, (size_t)128 * HW224);
  run_conv(c, img1, ehi.c, nullptr, f1c, 3, 224, 224, 128, 3, 3, 1, 1);
  float* f1r = res_block(c, f1c, 128, 128, 224, 224, ehi.r);
  float* feat1 = down_block(c, f1r, 128, 224, 224, ehi.d);       // 128@112

  // ---- get_texture ----
  float* flowh = wsalloc(c, (size_t)4 * HW112);                  // resize_half * 0.5
  avgpool2_scale_kernel<<<cdiv_h(4 * HW112, 256), 256, 0, stream>>>(flow, flowh, 4, 224, 224, 0.5f);
  // forward_to_backward over batch-2 flow_cat (memory layout already (2,2,112,112))
  float* acc2 = wsalloc(c, (size_t)2 * 3 * HW112);
  run_fill(c, acc2, (long)2 * 3 * HW112, 0.0f);
  softsplat_scatter_kernel<<<cdiv_h(2 * HW112, 256), 256, 0, stream>>>(flowh, flowh, acc2, 2, 2, 112, 112);
  float* bflow = wsalloc(c, (size_t)4 * HW112);
  softsplat_norm_kernel<<<cdiv_h(4 * HW112, 256), 256, 0, stream>>>(acc2, bflow, 2, 2, HW112, -1.0f);
  float* f0s = wsalloc(c, (size_t)2 * 56 * 56);
  float* f1s = wsalloc(c, (size_t)2 * 56 * 56);
  subsample2_scale_kernel<<<cdiv_h(2 * 56 * 56, 256), 256, 0, stream>>>(bflow, f0s, 2, 112, 112, 0.5f);
  subsample2_scale_kernel<<<cdiv_h(2 * 56 * 56, 256), 256, 0, stream>>>(bflow + (long)2 * HW112, f1s, 2, 112, 112, 0.5f);

  float* k0w = wsalloc(c, (size_t)128 * 56 * 56);
  float* k1w = wsalloc(c, (size_t)128 * 56 * 56);
  warp_nearest_kernel<<<cdiv_h(56 * 56, 256), 256, 0, stream>>>(feat0, f0s, k0w, 128, 112, 112, 56, 56);
  warp_nearest_kernel<<<cdiv_h(56 * 56, 256), 256, 0, stream>>>(feat1, f1s, k1w, 128, 112, 112, 56, 56);

  float* v0f = wsalloc(c, (size_t)128 * HW112);
  float* v1f = wsalloc(c, (size_t)128 * HW112);
  warp_unfold_kernel<<<cdiv_h(56 * 56, 256), 256, 0, stream>>>(feat0, f0s, c.sB, 128, 112, 112, 56, 56);
  fold2_kernel<<<cdiv_h(128 * HW112, 256), 256, 0, stream>>>(c.sB, v0f, 128, 56, 56);
  warp_unfold_kernel<<<cdiv_h(56 * 56, 256), 256, 0, stream>>>(feat1, f1s, c.sB, 128, 112, 112, 56, 56);
  fold2_kernel<<<cdiv_h(128 * HW112, 256), 256, 0, stream>>>(c.sB, v1f, 128, 56, 56);

  const int L = 28 * 28;
  float* sQ = wsalloc(c, (size_t)L * 128);
  tex_sq_kernel<<<L, 128, 0, stream>>>(Q, sQ, 128, 56, 56);
  float* sK = wsalloc(c, (size_t)2 * 128 * L);
  tex_sk_kernel<<<dim3(L, 2), 128, 0, stream>>>(k0w, k1w, sK, 128, 56, 56);
  int* best = (int*)wsalloc(c, (size_t)L);
  tex_argmax_kernel<<<L, 128, 0, stream>>>(sK, sQ, best, 128, 28, 28);
  float* texture = wsalloc(c, (size_t)128 * HW112);
  tex_gather_kernel<<<cdiv_h(128 * HW112, 256), 256, 0, stream>>>(best, v0f, v1f, texture, 128, 28, 112);

  // ---- merge + main U-Net ----
  float* m = cat2(c, wfeat, 128, texture, 128, HW112);           // 256@112
  float* mm = res_block(c, m, 256, 256, 112, 112, mf1);
  mm = res_block(c, mm, 256, 128, 112, 112, mf2);
  mm = res_block(c, mm, 128, 128, 112, 112, mf3);

  unet_run(c, mm, U, (float*)d_out);
}
